// ProximalAttention2D_PyTorch_66331474919552
// MI455X (gfx1250) — matrix operand order repaired and checked by reading, not compiled, not run on hardware
//
#include <hip/hip_runtime.h>
#include <hip/hip_fp16.h>

#define EMBED    256
#define HEADS    8
#define HEAD_DIM 32
#define KS       7
#define PADR     3
#define BB       4
#define HH       56
#define WWID     56
#define NPOS     (BB * HH * WWID)   // 12544

#define LDS_ROW  264                // 256 f16 + 8 f16 pad -> 528 B row stride (bank skew 4)

typedef __attribute__((ext_vector_type(16))) _Float16 v16h;
typedef __attribute__((ext_vector_type(8)))  _Float16 v8h;
typedef __attribute__((ext_vector_type(8)))  float    v8f;
typedef int v4i_vs __attribute__((vector_size(4 * sizeof(int))));   // matches builtin proto

#if defined(__gfx1250__) && __has_builtin(__builtin_amdgcn_global_load_async_to_lds_b128)
#define USE_ASYNC_LDS 1
#else
#define USE_ASYNC_LDS 0
#endif

// ---------------------------------------------------------------- f32 -> f16
__global__ __launch_bounds__(256) void cvt_f32_f16(const float* __restrict__ src,
                                                   __half* __restrict__ dst, int n) {
    int i = blockIdx.x * blockDim.x + threadIdx.x;
    int stride = gridDim.x * blockDim.x;
    for (; i < n; i += stride) dst[i] = __float2half(src[i]);
}

// -------------------------------------------------- C[M,256] = A[M,256] @ W[256,256]^T + bias
// Block = 128 threads (4 waves). The block's 16-row A tile is staged to LDS once
// (async global->LDS when available), then each wave computes one 16x16 C tile:
// 8x v_wmma_f32_16x16x32_f16 over K=256, A fragments from LDS, B fragments from
// global (W row-major == B column-major, 16 contiguous K per lane-half).
__global__ __launch_bounds__(128) void wmma_gemm_bias(const __half* __restrict__ Ah,
                                                      const __half* __restrict__ Wh,
                                                      const float* __restrict__ bias,
                                                      float* __restrict__ C) {
    const _Float16* A = reinterpret_cast<const _Float16*>(Ah);
    const _Float16* W = reinterpret_cast<const _Float16*>(Wh);

    __shared__ __align__(16) _Float16 atile[16 * LDS_ROW];   // 8448 B

    const int tid     = threadIdx.x;
    const int lane    = tid & 31;
    const int wave    = tid >> 5;
    const int m0      = blockIdx.x * 16;
    const int n0      = (blockIdx.y * 4 + wave) * 16;
    const int half_hi = lane >> 4;             // 0: lanes 0-15, 1: lanes 16-31
    const int cB      = n0 + (lane & 15);      // W row == B column == output column

    // ---- stage A tile: 16 rows x 512 B = 512 x 16B chunks, 4 per thread ----
#pragma unroll
    for (int i = 0; i < 4; ++i) {
        const int c   = tid + i * 128;         // chunk id 0..511
        const int row = c >> 5;                // 32 chunks of 16 B per row
        const int col = (c & 31) * 8;          // f16 units
        const _Float16* g = A + (size_t)(m0 + row) * EMBED + col;
        _Float16*       l = &atile[row * LDS_ROW + col];
#if USE_ASYNC_LDS
        __builtin_amdgcn_global_load_async_to_lds_b128((v4i_vs*)g, (v4i_vs*)l, 0, 0);
#else
        *(v8h*)l = *(const v8h*)g;
#endif
    }
#if USE_ASYNC_LDS
#if __has_builtin(__builtin_amdgcn_s_wait_asynccnt)
    __builtin_amdgcn_s_wait_asynccnt(0);
#else
    asm volatile("s_wait_asynccnt 0x0" ::: "memory");
#endif
#endif
    __syncthreads();

    // ---- WMMA main loop ----
    const _Float16* aRow = &atile[(lane & 15) * LDS_ROW];   // A row for this lane
    const _Float16* bRow = W + (size_t)cB * EMBED;
    const int aShift = half_hi * 16;
    const int bShift = half_hi * 16;

    v8f c = {};
#pragma unroll
    for (int k0 = 0; k0 < EMBED; k0 += 32) {
        v8h alo = *(const v8h*)(aRow + k0 + aShift);        // ds_load_b128
        v8h ahi = *(const v8h*)(aRow + k0 + 8 + aShift);    // ds_load_b128
        v16h a;
#pragma unroll
        for (int i = 0; i < 8; ++i) { a[i] = alo[i]; a[8 + i] = ahi[i]; }
        v16h b = *(const v16h*)(bRow + k0 + bShift);        // 2x global_load_b128
        // (neg_a, A, neg_b, B, c_mod, C, reuse_a, reuse_b)
        c = __builtin_amdgcn_wmma_f32_16x16x32_f16(false, a, false, b, (short)0, c,
                                                   false, false);
    }

    const float bv = bias[cB];
#pragma unroll
    for (int r = 0; r < 8; ++r) {
        const int m = m0 + r + half_hi * 8;    // C/D layout: VGPR r -> row r (+8 hi half)
        C[(size_t)m * EMBED + cB] = c[r] + bv;
    }
}

// -------------------------------------------------- 7x7 neighborhood attention, online softmax
// One thread per (position, head). Zero padding: OOB neighbor -> logit 0 (kept in
// softmax denominator) and zero v contribution — faithful to the reference.
__global__ __launch_bounds__(256) void prox_attn(const float* __restrict__ q,
                                                 const float* __restrict__ k,
                                                 const float* __restrict__ v,
                                                 __half* __restrict__ o) {
    int idx = blockIdx.x * blockDim.x + threadIdx.x;       // over NPOS*HEADS
    if (idx >= NPOS * HEADS) return;
    const int head = idx & (HEADS - 1);
    const int pos  = idx >> 3;
    const int wcol = pos % WWID;
    const int hrow = (pos / WWID) % HH;
    const int bimg = pos / (WWID * HH);
    const float scale = 0.17677669529663687f;              // 1/sqrt(32)

    const float* qp = q + (size_t)pos * EMBED + head * HEAD_DIM;
    float qr[HEAD_DIM];
#pragma unroll
    for (int i = 0; i < HEAD_DIM / 4; ++i) {
        float4 t = ((const float4*)qp)[i];
        qr[4 * i] = t.x; qr[4 * i + 1] = t.y; qr[4 * i + 2] = t.z; qr[4 * i + 3] = t.w;
    }

    float mx = -3.4e38f, sum = 0.f;
    float acc[HEAD_DIM];
#pragma unroll
    for (int i = 0; i < HEAD_DIM; ++i) acc[i] = 0.f;

    for (int di = 0; di < KS; ++di) {
        for (int dj = 0; dj < KS; ++dj) {
            const int hh = hrow + di - PADR;
            const int ww = wcol + dj - PADR;
            const bool inb = (hh >= 0) && (hh < HH) && (ww >= 0) && (ww < WWID);
            size_t nbr = 0;
            float logit = 0.f;
            if (inb) {
                nbr = ((size_t)(bimg * HH + hh) * WWID + ww) * EMBED + head * HEAD_DIM;
                const float4* kp = (const float4*)(k + nbr);
                float d0 = 0.f;
#pragma unroll
                for (int i = 0; i < HEAD_DIM / 4; ++i) {
                    float4 t = kp[i];
                    d0 = fmaf(qr[4 * i], t.x,
                         fmaf(qr[4 * i + 1], t.y,
                         fmaf(qr[4 * i + 2], t.z,
                         fmaf(qr[4 * i + 3], t.w, d0))));
                }
                logit = d0 * scale;
            }
            const float mnew = fmaxf(mx, logit);
            const float corr = __expf(mx - mnew);
            const float wgt  = __expf(logit - mnew);
            sum = sum * corr + wgt;
#pragma unroll
            for (int i = 0; i < HEAD_DIM; ++i) acc[i] *= corr;
            if (inb) {
                const float4* vp = (const float4*)(v + nbr);
#pragma unroll
                for (int i = 0; i < HEAD_DIM / 4; ++i) {
                    float4 t = vp[i];
                    acc[4 * i]     = fmaf(wgt, t.x, acc[4 * i]);
                    acc[4 * i + 1] = fmaf(wgt, t.y, acc[4 * i + 1]);
                    acc[4 * i + 2] = fmaf(wgt, t.z, acc[4 * i + 2]);
                    acc[4 * i + 3] = fmaf(wgt, t.w, acc[4 * i + 3]);
                }
            }
            mx = mnew;
        }
    }
    const float inv = 1.f / sum;
    __half* op = o + (size_t)pos * EMBED + head * HEAD_DIM;
#pragma unroll
    for (int i = 0; i < HEAD_DIM; ++i) op[i] = __float2half(acc[i] * inv);
}

// ----------------------------------------------------------------------------
extern "C" void kernel_launch(void* const* d_in, const int* in_sizes, int n_in,
                              void* d_out, int out_size, void* d_ws, size_t ws_size,
                              hipStream_t stream) {
    const float* x  = (const float*)d_in[0];
    const float* wq = (const float*)d_in[1];
    const float* wk = (const float*)d_in[2];
    const float* wv = (const float*)d_in[3];
    const float* wo = (const float*)d_in[4];
    const float* bq = (const float*)d_in[5];
    const float* bk = (const float*)d_in[6];
    const float* bv = (const float*)d_in[7];
    const float* bo = (const float*)d_in[8];
    float* out = (float*)d_out;

    char* ws = (char*)d_ws;
    size_t off = 0;
    __half* x16  = (__half*)(ws + off); off += (size_t)NPOS * EMBED * sizeof(__half);
    __half* wq16 = (__half*)(ws + off); off += (size_t)EMBED * EMBED * sizeof(__half);
    __half* wk16 = (__half*)(ws + off); off += (size_t)EMBED * EMBED * sizeof(__half);
    __half* wv16 = (__half*)(ws + off); off += (size_t)EMBED * EMBED * sizeof(__half);
    __half* wo16 = (__half*)(ws + off); off += (size_t)EMBED * EMBED * sizeof(__half);
    float*  qf   = (float*)(ws + off);  off += (size_t)NPOS * EMBED * sizeof(float);
    float*  kf   = (float*)(ws + off);  off += (size_t)NPOS * EMBED * sizeof(float);
    float*  vf   = (float*)(ws + off);  off += (size_t)NPOS * EMBED * sizeof(float);
    __half* a16  = (__half*)(ws + off); off += (size_t)NPOS * EMBED * sizeof(__half);

    // Stage 1: f32 -> f16 staging of x and weights.
    const int nX = NPOS * EMBED;          // 3,211,264
    const int nW = EMBED * EMBED;         // 65,536
    cvt_f32_f16<<<(nX + 255) / 256, 256, 0, stream>>>(x,  x16,  nX);
    cvt_f32_f16<<<(nW + 255) / 256, 256, 0, stream>>>(wq, wq16, nW);
    cvt_f32_f16<<<(nW + 255) / 256, 256, 0, stream>>>(wk, wk16, nW);
    cvt_f32_f16<<<(nW + 255) / 256, 256, 0, stream>>>(wv, wv16, nW);
    cvt_f32_f16<<<(nW + 255) / 256, 256, 0, stream>>>(wo, wo16, nW);

    // Stage 2: q/k/v projections (WMMA f16 -> f32).
    dim3 ggrid(NPOS / 16, EMBED / 16 / 4);   // (784, 4), 4 waves/block over N-tiles
    dim3 gblk(128);
    wmma_gemm_bias<<<ggrid, gblk, 0, stream>>>(x16, wq16, bq, qf);
    wmma_gemm_bias<<<ggrid, gblk, 0, stream>>>(x16, wk16, bk, kf);
    wmma_gemm_bias<<<ggrid, gblk, 0, stream>>>(x16, wv16, bv, vf);

    // Stage 3: 7x7 neighborhood attention with online softmax.
    const int nAttn = NPOS * HEADS;       // 100,352
    prox_attn<<<(nAttn + 255) / 256, 256, 0, stream>>>(qf, kf, vf, a16);

    // Stage 4: output projection (WMMA) into d_out (f32).
    wmma_gemm_bias<<<ggrid, gblk, 0, stream>>>(a16, wo16, bo, out);
}
